// MultiRegionRNN_23622320128402
// MI455X (gfx1250) — matrix operand order repaired and checked by reading, not compiled, not run on hardware
//
#include <hip/hip_runtime.h>
#include <math.h>

typedef unsigned short u16;
typedef __attribute__((ext_vector_type(16))) __bf16 v16bf;
typedef __attribute__((ext_vector_type(8)))  float  v8f;

// ---- problem constants (match reference) ----
constexpr int   T_STEPS = 512;
constexpr int   BATCH   = 128;
constexpr int   D_IN    = 64;
constexpr int   NR      = 512;
constexpr int   D_OUT   = 2;
constexpr float ALPHA_F = 0.1f;

constexpr int NBLK = 96;             // 3 regions * 32 n-columns; must all be co-resident
constexpr int KA   = NR + NR + D_IN; // 1088 : [W_rec_A | W_CA | W_in_A]
constexpr int KB   = NR + NR;        // 1024 : [W_rec | W_cross]

// ---- workspace layout (bytes) ----
constexpr size_t OFF_CNT   = 0;                                   // barrier counter
constexpr size_t OFF_RATES = 256;                                 // u16[2][3][B][N] bf16 rates
constexpr size_t SZ_RATES  = (size_t)2 * 3 * BATCH * NR * 2;      // 786432
constexpr size_t OFF_RCF   = OFF_RATES + SZ_RATES;                // f32 rC for readout [2][B][N]
constexpr size_t SZ_RCF    = (size_t)2 * BATCH * NR * 4;          // 524288
constexpr size_t OFF_WA    = OFF_RCF + SZ_RCF;                    // bf16 WA [512][1088]
constexpr size_t SZ_WA     = (size_t)NR * KA * 2;
constexpr size_t OFF_WB    = OFF_WA + SZ_WA;                      // bf16 WB [512][1024]
constexpr size_t SZ_WB     = (size_t)NR * KB * 2;
constexpr size_t OFF_WC    = OFF_WB + SZ_WB;                      // bf16 WC [512][1024]
constexpr size_t OFF_X     = OFF_WC + SZ_WB;                      // bf16 X  [512][128][64]
constexpr size_t SZ_X      = (size_t)T_STEPS * BATCH * D_IN * 2;
constexpr size_t WS_NEED   = OFF_X + SZ_X;                        // ~12.3 MB

__device__ __forceinline__ u16 f2bf(float f) {
    unsigned u = __float_as_uint(f);
    unsigned r = (u + 0x7FFFu + ((u >> 16) & 1u)) >> 16;  // round-to-nearest-even
    return (u16)r;
}

// ---- init kernels: re-run every launch (deterministic; counter MUST restart at 0) ----
__global__ void k_zero(unsigned* __restrict__ p, int n) {
    int i = blockIdx.x * blockDim.x + threadIdx.x;
    if (i < n) p[i] = 0u;
}

__global__ void k_convw(const float* __restrict__ WrA, const float* __restrict__ WrB,
                        const float* __restrict__ WrC, const float* __restrict__ WAB,
                        const float* __restrict__ WBC, const float* __restrict__ WCA,
                        const float* __restrict__ WinA,
                        u16* __restrict__ WA, u16* __restrict__ WB, u16* __restrict__ WC) {
    const int EA = NR * KA;            // 557056
    const int EB = NR * KB;            // 524288
    int i = blockIdx.x * blockDim.x + threadIdx.x;
    if (i < EA) {
        int n = i / KA, k = i % KA;
        float f = (k < NR)     ? WrA[n * NR + k]
                : (k < 2 * NR) ? WCA[n * NR + (k - NR)]
                               : WinA[n * D_IN + (k - 2 * NR)];
        WA[i] = f2bf(f);
    } else if (i < EA + EB) {
        int j = i - EA;
        int n = j / KB, k = j % KB;
        float f = (k < NR) ? WrB[n * NR + k] : WAB[n * NR + (k - NR)];
        WB[j] = f2bf(f);
    } else if (i < EA + 2 * EB) {
        int j = i - EA - EB;
        int n = j / KB, k = j % KB;
        float f = (k < NR) ? WrC[n * NR + k] : WBC[n * NR + (k - NR)];
        WC[j] = f2bf(f);
    }
}

__global__ void k_convx(const float* __restrict__ X, u16* __restrict__ Xb, int n) {
    int i = blockIdx.x * blockDim.x + threadIdx.x;
    if (i < n) Xb[i] = f2bf(X[i]);
}

union FragBF { v16bf v; uint4 q[2]; };

// Branch-free K segment: NCHUNK chunks of K=32. `ap` (global, per-lane) and
// `lp` (LDS, per-lane) are loop-invariant; chunk offsets fold into the
// instruction-immediate offset fields of global_load_b128 / ds_load_b128.
template <int NCHUNK>
__device__ __forceinline__ v8f kseg(const u16* __restrict__ ap, const u16* lp, v8f acc) {
#pragma unroll 8
    for (int kc = 0; kc < NCHUNK; ++kc) {
        FragBF fa, fb;
        // A-frag (16-bit 16x32): lane holds row; K = kc*32 + agrp*8 + {0..7, 16..23}
        fa.q[0] = *(const uint4*)(ap + kc * 32);
        fa.q[1] = *(const uint4*)(ap + kc * 32 + 16);
        // B-frag (32x16): lane holds column; K = kc*32 + agrp*8 + {0..7, 16..23}
        fb.q[0] = *(const uint4*)(lp + kc * 32);
        fb.q[1] = *(const uint4*)(lp + kc * 32 + 16);
        acc = __builtin_amdgcn_wmma_f32_16x16x32_bf16(false, fa.v, false, fb.v,
                                                      (short)0, acc, false, false);
    }
    return acc;
}

// ---- persistent RNN kernel: 96 blocks * 256 threads (8 wave32) ----
// block g: region = g/32 (A,B,C), ncol = g%32 (16-wide neuron column)
// wave w:  16x16 output tile over batch rows [16w, 16w+16)
__global__ __launch_bounds__(256, 1)
void rnn_persistent(const u16* __restrict__ WAc, const u16* __restrict__ WBc,
                    const u16* __restrict__ WCc, const u16* __restrict__ Xbf,
                    const float* __restrict__ WoutC,
                    u16* __restrict__ rates, float* __restrict__ rcf,
                    unsigned* __restrict__ cnt, float* __restrict__ out) {
    __shared__ u16 smW[16 * KA];  // this block's 16 weight rows (bf16), row stride K

    const int tid    = threadIdx.x;
    const int lane   = tid & 31;
    const int wv     = tid >> 5;
    const int region = blockIdx.x >> 5;
    const int ncol   = blockIdx.x & 31;
    const int K      = (region == 0) ? KA : KB;
    const u16* Wsrc  = (region == 0) ? WAc : (region == 1) ? WBc : WCc;

    // Stage weights into LDS once (constant across all 511 steps).
    {
        const uint4* src = (const uint4*)(Wsrc + (size_t)(ncol * 16) * K);
        uint4* dst = (uint4*)smW;
        const int n16 = K * 2;  // (16*K u16 * 2B) / 16B
        for (int i = tid; i < n16; i += 256) dst[i] = src[i];
    }
    __syncthreads();

    const int mrow = wv * 16 + (lane & 15);  // batch row this lane supplies to A-frag
    const int agrp = lane >> 4;              // K-group select (ISA 16-bit A layout)
    const int bn   = lane & 15;              // tile column (B/C/D layout: N = lane%16)
    const int nglb = ncol * 16 + bn;

    // loop-invariant per-lane offsets
    const size_t aoff  = (size_t)mrow * NR + agrp * 8;        // into rate buffers
    const size_t xoff  = (size_t)mrow * D_IN + agrp * 8;      // into X[t]
    const u16*   lbase = smW + (size_t)bn * K + agrp * 8;     // LDS weight base

    // which two rate segments feed this region: A<-[rA|rC|x], B<-[rB|rA], C<-[rC|rB]
    const int seg0 = (region == 0) ? 0 : (region == 1) ? 1 : 2;
    const int seg1 = (region == 0) ? 2 : (region == 1) ? 0 : 1;

    float xst[8];
#pragma unroll
    for (int v = 0; v < 8; ++v) xst[v] = 0.0f;  // x=0, r=tanh(0)=0 at t=0

    for (int t = 1; t < T_STEPS; ++t) {
        const int cur = (t + 1) & 1;  // rates produced at t-1
        const int nxt = t & 1;        // rates produced at t
        const u16* rb = rates + (size_t)cur * (3 * BATCH * NR);

        v8f acc = {0.f, 0.f, 0.f, 0.f, 0.f, 0.f, 0.f, 0.f};
        acc = kseg<16>(rb + (size_t)seg0 * BATCH * NR + aoff, lbase, acc);
        acc = kseg<16>(rb + (size_t)seg1 * BATCH * NR + aoff, lbase + 512, acc);
        if (region == 0)  // external input x_t (K=64)
            acc = kseg<2>(Xbf + (size_t)t * BATCH * D_IN + xoff, lbase + 1024, acc);

        // leaky update + tanh; write new rates (bf16) for next step
        u16*   wr = rates + (size_t)nxt * (3 * BATCH * NR) + (size_t)region * BATCH * NR;
        float* wf = rcf + (size_t)nxt * (BATCH * NR);
#pragma unroll
        for (int v = 0; v < 8; ++v) {
            float xn = (1.0f - ALPHA_F) * xst[v] + ALPHA_F * acc[v];
            xst[v] = xn;
            float r = tanhf(xn);
            int brow = wv * 16 + v + 8 * agrp;  // C/D layout: M = v + 8*(lane/16)
            wr[(size_t)brow * NR + nglb] = f2bf(r);
            if (region == 2) wf[(size_t)brow * NR + nglb] = r;  // fp32 rC for readout
        }

        // ---- device-wide barrier (96 resident blocks, monotonic counter) ----
        __threadfence();
        __syncthreads();
        if (tid == 0) {
            atomicAdd(cnt, 1u);
            const unsigned target = (unsigned)(NBLK * t);
            volatile unsigned* vc = cnt;
            while (*vc < target) __builtin_amdgcn_s_sleep(2);
        }
        __syncthreads();
        __threadfence();

        // readout out_t = rC @ W_out_C.T  (block 0 only; others run ahead into t+1,
        // which is safe: they read buf[nxt] too and write buf[cur])
        if (blockIdx.x == 0) {
            const int b = tid >> 1, o = tid & 1;
            const float* rc = rcf + (size_t)nxt * (BATCH * NR) + (size_t)b * NR;
            const float* wo = WoutC + o * NR;
            float s = 0.0f;
#pragma unroll 4
            for (int k = 0; k < NR; k += 4) {
                float4 a = *(const float4*)(rc + k);
                float4 w = *(const float4*)(wo + k);
                s += a.x * w.x + a.y * w.y + a.z * w.z + a.w * w.w;
            }
            out[(size_t)(t - 1) * BATCH * D_OUT + b * D_OUT + o] = s;
        }
    }
}

extern "C" void kernel_launch(void* const* d_in, const int* in_sizes, int n_in,
                              void* d_out, int out_size, void* d_ws, size_t ws_size,
                              hipStream_t stream) {
    (void)in_sizes; (void)n_in; (void)out_size;
    if (ws_size < WS_NEED) return;  // workspace too small to run safely

    const float* X    = (const float*)d_in[0];
    const float* WrA  = (const float*)d_in[1];
    const float* WrB  = (const float*)d_in[2];
    const float* WrC  = (const float*)d_in[3];
    const float* WAB  = (const float*)d_in[4];
    const float* WBC  = (const float*)d_in[5];
    const float* WCA  = (const float*)d_in[6];
    const float* WinA = (const float*)d_in[7];
    const float* WoC  = (const float*)d_in[8];

    char* ws = (char*)d_ws;
    unsigned* cnt  = (unsigned*)(ws + OFF_CNT);
    u16*  rates    = (u16*)(ws + OFF_RATES);
    float* rcf     = (float*)(ws + OFF_RCF);
    u16*  WA       = (u16*)(ws + OFF_WA);
    u16*  WB       = (u16*)(ws + OFF_WB);
    u16*  WC       = (u16*)(ws + OFF_WC);
    u16*  Xb       = (u16*)(ws + OFF_X);

    // zero counter + both rate buffers (rates at t=0 are tanh(0)=0)
    const int nz = (int)((OFF_RATES + SZ_RATES) / 4);
    k_zero<<<(nz + 255) / 256, 256, 0, stream>>>((unsigned*)ws, nz);

    // pack + convert weights to bf16 concatenated layouts
    k_convw<<<(NR * KA + 2 * NR * KB) / 256, 256, 0, stream>>>(
        WrA, WrB, WrC, WAB, WBC, WCA, WinA, WA, WB, WC);

    // convert input sequence to bf16
    const int nx = T_STEPS * BATCH * D_IN;
    k_convx<<<nx / 256, 256, 0, stream>>>(X, Xb, nx);

    // persistent time-stepped RNN
    rnn_persistent<<<NBLK, 256, 0, stream>>>(WA, WB, WC, Xb, WoC,
                                             rates, rcf, cnt, (float*)d_out);
}
